// LinearAttention_26079041421695
// MI455X (gfx1250) — matrix operand order repaired and checked by reading, not compiled, not run on hardware
//
#include <hip/hip_runtime.h>
#include <math.h>

// ---------------------------------------------------------------------------
// LinearAttention on gfx1250 (MI455X) — f16 WMMA (16x16x32, f32 accumulate).
// x:[8,16,32,32,128]  w_qkv:[128,1536]  w_out:[512,128]  b_out/gamma/beta:[128]
//
// reshape(b, 8, n, 64) of [b,d,h,w,512] maps head = voxel_pos>>11 (contiguous
// 2048-voxel chunks); each voxel's 512 channels -> 8 rows of 64 within that
// head. context[b,h] (64x64) reduces over one 2048-voxel chunk.
//
// WMMA f16 operand layouts (wave32, per ISA 7.12.2):
//   A (16x32): lane row m=lane&15; half h -> K = (h>>3)*16 + (lane>>4)*8 + (h&7)
//   B (32x16): lane col n=lane&15; half h -> K as for A; this code uses K = (lane>>4)*16 + h for A and B
//   C/D (16x16 f32): VGPR r -> M = r + 8*(lane>>4), N = lane&15
// ---------------------------------------------------------------------------

typedef _Float16 v16h __attribute__((ext_vector_type(16)));
typedef _Float16 v8h  __attribute__((ext_vector_type(8)));
typedef float    v8f  __attribute__((ext_vector_type(8)));

#define C_IN   128
#define F_QKV  1536
#define NBLK   128      // voxels per tile
#define NTHR   256      // 8 waves (wave32)
#define NCHUNK 64       // (b * heads) chunks
#define NPART  4        // deterministic partial-context split per chunk

__device__ __forceinline__ v8f wmma16(v16h a, v16h b, v8f c) {
  return __builtin_amdgcn_wmma_f32_16x16x32_f16(false, a, false, b,
                                                (short)0, c, false, false);
}

// A operand: one contiguous 16-half run (K = base..base+15)
__device__ __forceinline__ v16h loadA(const _Float16* p0, const _Float16* p1) {
  v8h lo = *(const v8h*)p0;
  v8h hi = *(const v8h*)(p0 + 8);
  return __builtin_shufflevector(lo, hi, 0, 1, 2, 3, 4, 5, 6, 7,
                                 8, 9, 10, 11, 12, 13, 14, 15);
}

// ---------------------------------------------------------------------------
// Weight prep: convert to f16 and transpose to K-major (column-contiguous).
//   Wqkv_t[f][k] = Wqkv[k][f]   (1536 x 128)
//   Wout_t[c][e] = Wout[e][c]   (128 x 512)
// ---------------------------------------------------------------------------
__global__ void prep_weights_kernel(const float* __restrict__ Wqkv,
                                    const float* __restrict__ Wout,
                                    _Float16* __restrict__ Wqkv_t,
                                    _Float16* __restrict__ Wout_t) {
  const int i = blockIdx.x * blockDim.x + threadIdx.x;
  if (i < F_QKV * C_IN) {
    const int f = i >> 7, k = i & 127;
    Wqkv_t[i] = (_Float16)Wqkv[k * F_QKV + f];
  }
  if (i < C_IN * 512) {
    const int c = i >> 9, e = i & 511;
    Wout_t[i] = (_Float16)Wout[e * C_IN + c];
  }
}

// ---------------------------------------------------------------------------
// Pass 1: partial context accumulation. grid = NCHUNK*NPART WGs, 512 voxels/WG.
// ---------------------------------------------------------------------------
__global__ __launch_bounds__(NTHR) void ctx_accum_kernel(
    const float* __restrict__ X, const _Float16* __restrict__ Wqkv_t,
    float* __restrict__ ctx_part) {
  __shared__ alignas(32) _Float16 sXh[NBLK * C_IN];  // 32 KB, row-major
  __shared__ float                sK[NBLK * 64];     // 32 KB, f32 staging
  __shared__ alignas(32) _Float16 sKh[64 * NBLK];    // 16 KB, [c][row]
  __shared__ alignas(32) _Float16 sVh[64 * NBLK];    // 16 KB, [e][row]

  const int tid  = threadIdx.x;
  const int lane = tid & 31;
  const int wave = tid >> 5;
  const int mh   = lane >> 4;   // half-wave select
  const int ln   = lane & 15;
  const int koff = mh * 16;     // A-operand K sub-offset

  const int  chunk = blockIdx.x >> 2;
  const int  part  = blockIdx.x & 3;
  const long vbase = (long)chunk * 2048 + (long)part * 512;

  v8f accC[2];
  accC[0] = {}; accC[1] = {};

  for (int rb = 0; rb < 4; ++rb) {
    __syncthreads();
    const long voxel0 = vbase + (long)rb * NBLK;
    for (int i = tid; i < NBLK * C_IN; i += NTHR)
      sXh[i] = (_Float16)X[voxel0 * C_IN + i];
    __syncthreads();

    for (int j = 0; j < 8; ++j) {
      // ---- K,V GEMM: [128x128] x [128x64] (cols 512+64j / 1024+64j)
      v8f accK[4], accV[4];
#pragma unroll
      for (int t = 0; t < 4; ++t) { accK[t] = {}; accV[t] = {}; }
      const int kcol0 = 512 + 64 * j;
      const int vcol0 = 1024 + 64 * j;
      const int arow  = wave * 16 + ln;
      for (int k0 = 0; k0 < C_IN; k0 += 32) {
        const v16h a = loadA(&sXh[arow * C_IN + k0 + koff],
                             &sXh[arow * C_IN + k0 + 16 + koff]);
#pragma unroll
        for (int nt = 0; nt < 4; ++nt) {
          const v16h bk = *(const v16h*)(Wqkv_t +
              (size_t)(kcol0 + nt * 16 + ln) * C_IN + k0 + mh * 16);
          const v16h bv = *(const v16h*)(Wqkv_t +
              (size_t)(vcol0 + nt * 16 + ln) * C_IN + k0 + mh * 16);
          accK[nt] = wmma16(a, bk, accK[nt]);
          accV[nt] = wmma16(a, bv, accV[nt]);
        }
      }
      __syncthreads();  // prior j's sK/sKh/sVh consumers finished
#pragma unroll
      for (int nt = 0; nt < 4; ++nt)
#pragma unroll
        for (int r = 0; r < 8; ++r) {
          const int row = wave * 16 + r + 8 * mh;
          sK[row * 64 + nt * 16 + ln] = accK[nt][r];
          sVh[(nt * 16 + ln) * NBLK + row] = (_Float16)accV[nt][r];  // [e][row]
        }
      __syncthreads();

      // ---- softmax each K row (f32), emit transposed f16 [c][row]
      if (tid < NBLK) {
        float mx = -INFINITY;
        for (int c = 0; c < 64; ++c) mx = fmaxf(mx, sK[tid * 64 + c]);
        float s = 0.f;
        for (int c = 0; c < 64; ++c) {
          const float e = __expf(sK[tid * 64 + c] - mx);
          sK[tid * 64 + c] = e;
          s += e;
        }
        const float inv = 1.f / s;
        for (int c = 0; c < 64; ++c)
          sKh[c * NBLK + tid] = (_Float16)(sK[tid * 64 + c] * inv);
      }
      __syncthreads();

      // ---- context += K^T (64x128) * V (128x64); 16 tiles, 2 per wave
#pragma unroll
      for (int t = 0; t < 2; ++t) {
        const int tile = wave * 2 + t;
        const int cb = (tile >> 2) * 16;
        const int eb = (tile & 3) * 16;
        v8f acc = accC[t];
        for (int r0 = 0; r0 < NBLK; r0 += 32) {
          const v16h a = loadA(&sKh[(cb + ln) * NBLK + r0 + koff],
                               &sKh[(cb + ln) * NBLK + r0 + 16 + koff]);
          const v16h b = *(const v16h*)&sVh[(eb + ln) * NBLK + r0 + mh * 16];
          acc = wmma16(a, b, acc);
        }
        accC[t] = acc;
      }
    }
  }

  float* pbase = ctx_part + (long)blockIdx.x * 4096;
#pragma unroll
  for (int t = 0; t < 2; ++t) {
    const int tile = wave * 2 + t;
    const int cb = (tile >> 2) * 16;
    const int eb = (tile & 3) * 16;
#pragma unroll
    for (int r = 0; r < 8; ++r) {
      const int m = r + 8 * mh;
      pbase[(cb + m) * 64 + eb + ln] = accC[t][r];
    }
  }
}

// ---------------------------------------------------------------------------
// Deterministic fixed-order reduction of the NPART partial contexts.
// ---------------------------------------------------------------------------
__global__ void ctx_reduce_kernel(const float* __restrict__ parts,
                                  float* __restrict__ ctxF) {
  const int i = blockIdx.x * blockDim.x + threadIdx.x;  // < NCHUNK*4096
  if (i >= NCHUNK * 4096) return;
  const int chunk = i >> 12;
  const int off   = i & 4095;
  float s = 0.f;
#pragma unroll
  for (int p = 0; p < NPART; ++p)
    s += parts[((long)(chunk * NPART + p)) * 4096 + off];
  ctxF[i] = s;
}

// ---------------------------------------------------------------------------
// Pass 2: Q -> softmax*scale -> O=Q*ctx -> Y+=O*Wout -> bias+LayerNorm -> out
// ---------------------------------------------------------------------------
__global__ __launch_bounds__(NTHR) void out_kernel(
    const float* __restrict__ X, const _Float16* __restrict__ Wqkv_t,
    const float* __restrict__ ctx, const _Float16* __restrict__ Wout_t,
    const float* __restrict__ bout, const float* __restrict__ gamma,
    const float* __restrict__ beta, float* __restrict__ out) {
  __shared__ alignas(32) _Float16 sXh[NBLK * C_IN];  // 32 KB, row-major
  __shared__ alignas(32) _Float16 sCtxh[64 * 64];    //  8 KB, [e][d]
  __shared__ float                sQ[NBLK * 64];     // 32 KB, f32 staging
  __shared__ alignas(32) _Float16 sQh[NBLK * 64];    // 16 KB, row-major
  __shared__ alignas(32) _Float16 sOh[NBLK * 64];    // 16 KB, row-major
  __shared__ float                sY[NBLK * C_IN];   // 64 KB
  __shared__ float sMu[NBLK];
  __shared__ float sRs[NBLK];

  const int tid  = threadIdx.x;
  const int lane = tid & 31;
  const int wave = tid >> 5;
  const int mh   = lane >> 4;
  const int ln   = lane & 15;
  const int koff = mh * 16;

  const long voxel0 = (long)blockIdx.x * NBLK;
  const int  chunk  = (int)(voxel0 >> 11);

  for (int i = tid; i < NBLK * C_IN; i += NTHR)
    sXh[i] = (_Float16)X[voxel0 * C_IN + i];
  for (int i = tid; i < 4096; i += NTHR) {
    const int d = i >> 6, e = i & 63;
    sCtxh[e * 64 + d] = (_Float16)ctx[(long)chunk * 4096 + i];  // transpose
  }
  __syncthreads();

  v8f accY[8];
#pragma unroll
  for (int t = 0; t < 8; ++t) accY[t] = {};

  const float scale = 0.125f;  // DIM_HEAD^-0.5
  const int   arow  = wave * 16 + ln;

  for (int j = 0; j < 8; ++j) {
    // ---- Q GEMM: [128x128] x [128x64] (cols 64j..)
    v8f accQ[4];
#pragma unroll
    for (int t = 0; t < 4; ++t) accQ[t] = {};
    const int qcol0 = 64 * j;
    for (int k0 = 0; k0 < C_IN; k0 += 32) {
      const v16h a = loadA(&sXh[arow * C_IN + k0 + koff],
                           &sXh[arow * C_IN + k0 + 16 + koff]);
#pragma unroll
      for (int nt = 0; nt < 4; ++nt) {
        const v16h b = *(const v16h*)(Wqkv_t +
            (size_t)(qcol0 + nt * 16 + ln) * C_IN + k0 + mh * 16);
        accQ[nt] = wmma16(a, b, accQ[nt]);
      }
    }
    __syncthreads();  // prior O-GEMM (reads sQh) done
#pragma unroll
    for (int nt = 0; nt < 4; ++nt)
#pragma unroll
      for (int r = 0; r < 8; ++r) {
        const int row = wave * 16 + r + 8 * mh;
        sQ[row * 64 + nt * 16 + ln] = accQ[nt][r];
      }
    __syncthreads();

    // ---- softmax * scale per row (f32), emit row-major f16
    if (tid < NBLK) {
      float mx = -INFINITY;
      for (int c = 0; c < 64; ++c) mx = fmaxf(mx, sQ[tid * 64 + c]);
      float s = 0.f;
      for (int c = 0; c < 64; ++c) {
        const float e = __expf(sQ[tid * 64 + c] - mx);
        sQ[tid * 64 + c] = e;
        s += e;
      }
      const float inv = scale / s;
      for (int c = 0; c < 64; ++c)
        sQh[tid * 64 + c] = (_Float16)(sQ[tid * 64 + c] * inv);
    }
    __syncthreads();

    // ---- O = Q (128x64) * ctx (64x64)
    v8f accO[4];
#pragma unroll
    for (int t = 0; t < 4; ++t) accO[t] = {};
    for (int k0 = 0; k0 < 64; k0 += 32) {
      const v16h a = loadA(&sQh[arow * 64 + k0 + koff],
                           &sQh[arow * 64 + k0 + 16 + koff]);
#pragma unroll
      for (int nt = 0; nt < 4; ++nt) {
        const v16h b = *(const v16h*)&sCtxh[(nt * 16 + ln) * 64 + k0 + mh * 16];
        accO[nt] = wmma16(a, b, accO[nt]);
      }
    }
    __syncthreads();  // prior Y-GEMM (reads sOh) done
#pragma unroll
    for (int nt = 0; nt < 4; ++nt)
#pragma unroll
      for (int r = 0; r < 8; ++r) {
        const int row = wave * 16 + r + 8 * mh;
        sOh[row * 64 + nt * 16 + ln] = (_Float16)accO[nt][r];
      }
    __syncthreads();

    // ---- Y += O (128x64) * Wout[64j:64j+64, 0:128]   (B from Wout_t[c][e])
    for (int k0 = 0; k0 < 64; k0 += 32) {
      const v16h a = loadA(&sOh[arow * 64 + k0 + koff],
                           &sOh[arow * 64 + k0 + 16 + koff]);
#pragma unroll
      for (int nt = 0; nt < 8; ++nt) {
        const v16h b = *(const v16h*)(Wout_t +
            (size_t)(nt * 16 + ln) * 512 + 64 * j + k0 + mh * 16);
        accY[nt] = wmma16(a, b, accY[nt]);
      }
    }
  }

  __syncthreads();
#pragma unroll
  for (int nt = 0; nt < 8; ++nt)
#pragma unroll
    for (int r = 0; r < 8; ++r) {
      const int row = wave * 16 + r + 8 * mh;
      sY[row * C_IN + nt * 16 + ln] = accY[nt][r];
    }
  __syncthreads();

  // ---- bias + LayerNorm statistics (one thread per voxel row)
  if (tid < NBLK) {
    float s = 0.f, s2 = 0.f;
    for (int c = 0; c < C_IN; ++c) {
      const float v = sY[tid * C_IN + c] + bout[c];
      sY[tid * C_IN + c] = v;
      s += v;
      s2 += v * v;
    }
    const float mu  = s * (1.f / C_IN);
    const float var = s2 * (1.f / C_IN) - mu * mu;
    sMu[tid] = mu;
    sRs[tid] = rsqrtf(var + 1e-3f);
  }
  __syncthreads();

  for (int i = tid; i < NBLK * C_IN; i += NTHR) {
    const int r = i >> 7;
    const int c = i & 127;
    out[voxel0 * C_IN + i] = gamma[c] * (sY[i] - sMu[r]) * sRs[r] + beta[c];
  }
}

// ---------------------------------------------------------------------------
extern "C" void kernel_launch(void* const* d_in, const int* in_sizes, int n_in,
                              void* d_out, int out_size, void* d_ws, size_t ws_size,
                              hipStream_t stream) {
  const float* X     = (const float*)d_in[0];
  const float* Wqkv  = (const float*)d_in[1];
  const float* Wout  = (const float*)d_in[2];
  const float* bout  = (const float*)d_in[3];
  const float* gamma = (const float*)d_in[4];
  const float* beta  = (const float*)d_in[5];
  float* out = (float*)d_out;

  // ws layout: partial contexts (f32), reduced ctx (f32), f16 weights
  float*    parts  = (float*)d_ws;                                  // 1048576 f
  float*    ctxF   = parts + (size_t)NCHUNK * NPART * 4096;         //  262144 f
  _Float16* Wqkv_t = (_Float16*)(ctxF + (size_t)NCHUNK * 4096);     //  196608 h
  _Float16* Wout_t = Wqkv_t + (size_t)F_QKV * C_IN;                 //   65536 h

  const int nprep = F_QKV * C_IN;  // 196608 (covers Wout_t's 65536 too)
  prep_weights_kernel<<<(nprep + NTHR - 1) / NTHR, NTHR, 0, stream>>>(
      Wqkv, Wout, Wqkv_t, Wout_t);

  ctx_accum_kernel<<<NCHUNK * NPART, NTHR, 0, stream>>>(X, Wqkv_t, parts);

  const int nred = NCHUNK * 4096;
  ctx_reduce_kernel<<<(nred + NTHR - 1) / NTHR, NTHR, 0, stream>>>(parts, ctxF);

  out_kernel<<<1024, NTHR, 0, stream>>>(X, Wqkv_t, ctxF, Wout_t,
                                        bout, gamma, beta, out);
}
